// Gated_LSTM_16638703305446
// MI455X (gfx1250) — matrix operand order repaired and checked by reading, not compiled, not run on hardware
//
#include <hip/hip_runtime.h>

typedef __attribute__((ext_vector_type(16))) _Float16 v16h;
typedef __attribute__((ext_vector_type(8)))  _Float16 v8h;
typedef __attribute__((ext_vector_type(16))) float    v16f;
typedef __attribute__((ext_vector_type(8)))  float    v8f;
typedef __attribute__((ext_vector_type(4)))  unsigned int v4u;
typedef __attribute__((ext_vector_type(8)))  int      v8i;
typedef __attribute__((ext_vector_type(4)))  int      v4i;

#define Bn   64
#define Tn   2048
#define In   128
#define Hn   256
#define NWG  16
#define NTHR 512           // 16 waves of 32
// LDS row strides padded +16B to rotate banks (256B bank period / 64 banks)
#define HSTR 264           // halves  (528 B rows: 128 data dwords + 4 pad dwords)
#define XSTR 136           // halves  (272 B rows)
#define ZSTR 68            // floats  (272 B rows)

#ifndef __has_builtin
#define __has_builtin(x) 0
#endif
#if __has_builtin(__builtin_amdgcn_tensor_load_to_lds)
#define HAVE_TDM 1
#else
#define HAVE_TDM 0
#endif

__device__ __forceinline__ float fsig(float x) {
  return __builtin_amdgcn_rcpf(1.0f + __expf(-x));   // v_rcp_f32 instead of IEEE div
}
__device__ __forceinline__ float ftanh(float x) {
#if __has_builtin(__builtin_amdgcn_tanhf)
  return __builtin_amdgcn_tanhf(x);                  // V_TANH_F32 (gfx1250 TRANS op)
#else
  float e = __expf(x + x);
  return 1.0f - 2.0f * __builtin_amdgcn_rcpf(e + 1.0f);
#endif
}

__global__ void init_counter_k(unsigned int* c) {
  if (threadIdx.x == 0) *c = 0u;
}

// Persistent fused kernel: input gating + input GEMM + LSTM scan + output proj.
// 16 WGs x 512 threads; each wave owns one 16x16 tile of the 64x1024 step GEMM.
// Weight B-fragments (f16) stay in VGPRs for the whole kernel.
// 4H axis is permuted: col' = 4*j + gate (gate order i,f,g,o).
__global__ void __launch_bounds__(NTHR, 1)
gated_lstm_scan(const float* __restrict__ xd,   const float* __restrict__ w,
                const float* __restrict__ W_ih, const float* __restrict__ b_ih,
                const float* __restrict__ W_hh, const float* __restrict__ b_hh,
                const float* __restrict__ W_out,const float* __restrict__ b_out,
                float* __restrict__ out,
                unsigned int* __restrict__ counter,
                _Float16* __restrict__ h_gl)
{
  __shared__ _Float16 lds_h[Bn * HSTR];   // ~33 KB: full h (f16), padded rows
  __shared__ _Float16 lds_x[Bn * XSTR];   // ~17 KB: gated x_t (f16), padded rows
  __shared__ float    lds_z[Bn * ZSTR];   // ~17 KB: this WG's 64-col z slice (f32)

  const int tid    = threadIdx.x;
  const int wg     = blockIdx.x;
  const int wave   = tid >> 5;
  const int lane   = tid & 31;
  const int laneN  = lane & 15;
  const int hiHalf = lane >> 4;           // 0 or 1

  // ---- wave -> output tile mapping ----
  const int mtile = wave & 3;                        // 4 M tiles (M=64)
  const int ntile = wave >> 2;                       // 4 N tiles per WG
  const int colp  = wg * 64 + ntile * 16 + laneN;    // permuted column 0..1023
  const int orig  = (colp & 3) * Hn + (colp >> 2);   // original W_ih/W_hh row

  // ---- preload B fragments (weights) into VGPRs, f32 -> f16 ----
  v16h bh[8];  // W_hh, K = 0..255
  v16h bx[4];  // W_ih, K = 0..127
#pragma unroll
  for (int c = 0; c < 8; ++c) {
    const float* src = W_hh + (size_t)orig * Hn + 32 * c + hiHalf * 16;
    v16h bf;
#pragma unroll
    for (int k = 0; k < 16; ++k) bf[k] = (_Float16)src[k];
    bh[c] = bf;
  }
#pragma unroll
  for (int c = 0; c < 4; ++c) {
    const float* src = W_ih + (size_t)orig * In + 32 * c + hiHalf * 16;
    v16h bf;
#pragma unroll
    for (int k = 0; k < 16; ++k) bf[k] = (_Float16)src[k];
    bx[c] = bf;
  }

  // ---- elementwise mapping: thread -> 2 (batch, j) pairs, same j ----
  const int jl    = tid & 15;
  const int b0    = tid >> 4;              // 0..31
  const int b1    = b0 + 32;
  const int jglob = wg * 16 + jl;
  float bias_i = b_ih[0 * Hn + jglob] + b_hh[0 * Hn + jglob];
  float bias_f = b_ih[1 * Hn + jglob] + b_hh[1 * Hn + jglob];
  float bias_g = b_ih[2 * Hn + jglob] + b_hh[2 * Hn + jglob];
  float bias_o = b_ih[3 * Hn + jglob] + b_hh[3 * Hn + jglob];
  float cs0 = 0.0f, cs1 = 0.0f;            // cell state in f32 registers

  // ---- input-gating mapping: thread -> 16 contiguous inputs of one row ----
  const int    xrow  = tid >> 3;                      // batch 0..63
  const int    xi0   = (tid & 7) * 16;                // 0..112
  const size_t xbase = (size_t)xrow * Tn * In + xi0;

  // Phase A: gated = xd * sigmoid(w) -> lds_x (f16, padded rows)
  auto phaseA = [&](int tt) {
    v16f xv = *(const v16f*)(xd + xbase + (size_t)tt * In);
    v16f wv = *(const v16f*)(w  + xbase + (size_t)tt * In);
    if (tt + 1 < Tn) {                                 // global_prefetch_b8
      __builtin_prefetch(xd + xbase + (size_t)(tt + 1) * In, 0, 3);
      __builtin_prefetch(w  + xbase + (size_t)(tt + 1) * In, 0, 3);
    }
    v8h glo, ghi;
#pragma unroll
    for (int k = 0; k < 8; ++k) {
      glo[k] = (_Float16)(xv[k]     * fsig(wv[k]));
      ghi[k] = (_Float16)(xv[k + 8] * fsig(wv[k + 8]));
    }
    _Float16* dst = lds_x + xrow * XSTR + xi0;
    *(v8h*)dst       = glo;
    *(v8h*)(dst + 8) = ghi;
  };

#if HAVE_TDM
  // ---- Tensor Data Mover descriptor: 32KB h_gl -> lds_h, LDS-padded rows ----
  // D# group0: count=1 | lds_addr | global_addr[56:0] | type=2
  const unsigned           lds_h_addr = (unsigned)(size_t)(void*)lds_h;
  const unsigned long long ga         = (unsigned long long)(size_t)h_gl;
  const v4u tdm_g0 = { 1u,
                       lds_h_addr,
                       (unsigned)(ga & 0xFFFFFFFFu),
                       (unsigned)((ga >> 32) & 0x01FFFFFFu) | 0x80000000u };
  // group1: data_size=3 (8B) | pad_enable, pad_interval=6 (128 dw), pad_amount=3 (4 dw)
  //         tensor_dim0=4096, tensor_dim1=1, tile_dim0=4096, tile_dim1=1, stride0=4096
  const unsigned g1w0 = (3u << 16) | (1u << 20) | (6u << 22) | (3u << 25);
  const v8i tdm_g1 = { (int)g1w0,
                       (int)((4096u & 0xFFFFu) << 16),
                       (int)((4096u >> 16) | (1u << 16)),
                       (int)(4096u << 16),
                       1,
                       4096,
                       0, 0 };
  const v4i tdm_z4 = { 0, 0, 0, 0 };
#if __clang_major__ >= 23
  const v8i tdm_z8 = { 0, 0, 0, 0, 0, 0, 0, 0 };
#endif
#endif

  // ---- prologue: zero h, load x_0 ----
  for (int idx = tid; idx < (Bn * HSTR * 2) / 16; idx += NTHR)
    ((uint4*)lds_h)[idx] = make_uint4(0u, 0u, 0u, 0u);
  phaseA(0);

  const int arow = 16 * mtile + laneN;

  for (int t = 0; t < Tn; ++t) {
    __syncthreads();   // lds_x(t), lds_h(t) ready

    // -------- Phase B: z tile = h @ W_hh^T + gx @ W_ih^T (12 WMMAs) --------
    {
      const _Float16* hrow = lds_h + arow * HSTR;
      const _Float16* xrp  = lds_x + arow * XSTR;
      // Issue ALL 24 ds_load_b128 first, into 12 distinct register pairs, so
      // the WMMA chain runs behind partial s_wait_dscnt instead of full drains.
      // A-frag contiguous: lanes<16 hold K {0..15}, lanes>=16 {16..31}
      v8h alo[12], ahi[12];
#pragma unroll
      for (int c = 0; c < 12; ++c) {
        const _Float16* p = (c < 8) ? (hrow + 32 * c + hiHalf * 16)
                                    : (xrp  + 32 * (c - 8) + hiHalf * 16);
        alo[c] = *(const v8h*)p;
        ahi[c] = *(const v8h*)(p + 8);
      }
      v8f acc = {};
#pragma unroll
      for (int c = 0; c < 12; ++c) {
        v16h a = __builtin_shufflevector(alo[c], ahi[c],
                   0,1,2,3,4,5,6,7,8,9,10,11,12,13,14,15);
        acc = __builtin_amdgcn_wmma_f32_16x16x32_f16(
                  false, a, false, (c < 8) ? bh[c] : bx[c - 8],
                  (short)0, acc, false, false);
      }
      const int zrow0 = 16 * mtile + 8 * hiHalf;   // D-frag: M = r + 8*hiHalf
      const int zcol  = ntile * 16 + laneN;
#pragma unroll
      for (int r = 0; r < 8; ++r)
        lds_z[(zrow0 + r) * ZSTR + zcol] = acc[r];
    }
    __syncthreads();

    // -------- Phase C: gates, c/h update; publish h slice --------
    {
      float4 z0 = *(const float4*)(lds_z + b0 * ZSTR + 4 * jl);
      float4 z1 = *(const float4*)(lds_z + b1 * ZSTR + 4 * jl);
      float ig = fsig (z0.x + bias_i);
      float fg = fsig (z0.y + bias_f);
      float gg = ftanh(z0.z + bias_g);
      float og = fsig (z0.w + bias_o);
      cs0 = fg * cs0 + ig * gg;
      float h0 = og * ftanh(cs0);
      ig = fsig (z1.x + bias_i);
      fg = fsig (z1.y + bias_f);
      gg = ftanh(z1.z + bias_g);
      og = fsig (z1.w + bias_o);
      cs1 = fg * cs1 + ig * gg;
      float h1 = og * ftanh(cs1);
      h_gl[b0 * Hn + jglob] = (_Float16)h0;
      h_gl[b1 * Hn + jglob] = (_Float16)h1;
    }

    // overlap next step's gated-x load with the grid barrier below
    if (t + 1 < Tn) phaseA(t + 1);

    // -------- grid-wide step barrier (atomic release/acquire) --------
    __threadfence();
    __syncthreads();
    if (tid == 0) {
      __hip_atomic_fetch_add(counter, 1u, __ATOMIC_RELEASE, __HIP_MEMORY_SCOPE_AGENT);
      const unsigned int want = (unsigned int)(t + 1) * NWG;
      while (__hip_atomic_load(counter, __ATOMIC_ACQUIRE, __HIP_MEMORY_SCOPE_AGENT) < want)
        __builtin_amdgcn_s_sleep(1);
    }
    if (wave == 0)
      __builtin_amdgcn_s_cluster_barrier();   // s_barrier_signal/wait -3 (NOP unclustered)
    __syncthreads();
    __threadfence();

    // -------- Phase D: reload full h into LDS --------
#if HAVE_TDM
    if (wave == 0) {
#if __clang_major__ >= 23
      __builtin_amdgcn_tensor_load_to_lds(tdm_g0, tdm_g1, tdm_z4, tdm_z4, tdm_z8, 0);
#else
      __builtin_amdgcn_tensor_load_to_lds(tdm_g0, tdm_g1, tdm_z4, tdm_z4, 0);
#endif
      __builtin_amdgcn_s_wait_tensorcnt(0);
    }
#else
    for (int idx = tid; idx < Bn * Hn / 8; idx += NTHR) {
      const int row = idx >> 5, colq = idx & 31;     // 32 uint4 per 512B row
      *(uint4*)(lds_h + row * HSTR + colq * 8) = ((const uint4*)h_gl)[idx];
    }
#endif
    // loop-top __syncthreads makes the reload visible to all waves
  }

  // -------- output projection: out[b] = h_T[b,:] . W_out + b_out --------
  if (wg == 0 && tid < Bn) {
    float acc = b_out[0];
    const _Float16* hr = h_gl + tid * Hn;
#pragma unroll 8
    for (int k = 0; k < Hn; ++k)
      acc += (float)hr[k] * W_out[k];
    out[tid] = acc;
  }
}

extern "C" void kernel_launch(void* const* d_in, const int* in_sizes, int n_in,
                              void* d_out, int out_size, void* d_ws, size_t ws_size,
                              hipStream_t stream) {
  (void)in_sizes; (void)n_in; (void)out_size; (void)ws_size;
  const float* xd    = (const float*)d_in[0];
  const float* w     = (const float*)d_in[1];
  const float* W_ih  = (const float*)d_in[2];
  const float* b_ih  = (const float*)d_in[3];
  const float* W_hh  = (const float*)d_in[4];
  const float* b_hh  = (const float*)d_in[5];
  const float* W_out = (const float*)d_in[6];
  const float* b_out = (const float*)d_in[7];
  float* out = (float*)d_out;

  unsigned int* counter = (unsigned int*)d_ws;               // 4 B
  _Float16*     h_gl    = (_Float16*)((char*)d_ws + 128);    // 32 KB h buffer

  init_counter_k<<<1, 64, 0, stream>>>(counter);
  gated_lstm_scan<<<NWG, NTHR, 0, stream>>>(xd, w, W_ih, b_ih, W_hh, b_hh,
                                            W_out, b_out, out, counter, h_gl);
}
